// KernelGraphCalcLayer_68453188763813
// MI455X (gfx1250) — matrix operand order repaired and checked by reading, not compiled, not run on hardware
//
#include <hip/hip_runtime.h>
#include <hip/hip_bf16.h>

typedef __attribute__((ext_vector_type(16))) __bf16 v16bf;
typedef __attribute__((ext_vector_type(8)))  __bf16 v8bf;
typedef __attribute__((ext_vector_type(8)))  float  v8f;
typedef __attribute__((ext_vector_type(4)))  float  v4f;

#define BB   64
#define NN   256
#define IND  512
#define OUTD 512
#define KK   8
#define DKD  64

// ---------------------------------------------------------------------------
// Stage 0: wT[o][i] = (bf16) weight[i][o]   (512x512, tiny)
// ---------------------------------------------------------------------------
__global__ void __launch_bounds__(256)
kgc_transpose_w(const float* __restrict__ w, __bf16* __restrict__ wT) {
    int idx = blockIdx.x * 256 + threadIdx.x;   // coalesced read
    int i = idx >> 9;          // /512
    int o = idx & 511;
    wT[(size_t)o * IND + i] = (__bf16)w[idx];
}

// ---------------------------------------------------------------------------
// Stage 1: hT[b][o][n] = relu( X[b*N+n][:] . W[:][o] + bias[o] )  in bf16
// One wave computes one 16x16 output tile via 16 x v_wmma_f32_16x16x32_bf16.
// grid = (1024 m-tiles, 4), block = 256 (8 waves -> 8 n-tiles)
// ---------------------------------------------------------------------------
__global__ void __launch_bounds__(256)
kgc_linear_relu(const float* __restrict__ X, const __bf16* __restrict__ wT,
                const float* __restrict__ bias, __bf16* __restrict__ hT) {
    const int lane  = threadIdx.x & 31;
    const int wave  = threadIdx.x >> 5;
    const int mtile = blockIdx.x;               // 0..1023 over B*N/16
    const int ntile = blockIdx.y * 8 + wave;    // 0..31 over OUT/16
    const int half  = lane >> 4;                // 0/1
    const int l15   = lane & 15;

    const int mrow = mtile * 16 + l15;          // A row (global b*N+n)
    const int ocol = ntile * 16 + l15;          // B column (out dim)
    const float*  xrow = X  + (size_t)mrow * IND + half * 16;
    const __bf16* wrow = wT + (size_t)ocol * IND + half * 16;

    v8f c = {};
#pragma unroll 4
    for (int ks = 0; ks < IND / 32; ++ks) {
        v4f x0 = *(const v4f*)(xrow + ks * 32);
        v4f x1 = *(const v4f*)(xrow + ks * 32 + 4);
        v4f x2 = *(const v4f*)(xrow + ks * 32 + 8);
        v4f x3 = *(const v4f*)(xrow + ks * 32 + 12);
        v16bf a;
#pragma unroll
        for (int j = 0; j < 4; ++j) {
            a[j]      = (__bf16)x0[j];
            a[4 + j]  = (__bf16)x1[j];
            a[8 + j]  = (__bf16)x2[j];
            a[12 + j] = (__bf16)x3[j];
        }
        v16bf bfrag = *(const v16bf*)(wrow + ks * 32);
        c = __builtin_amdgcn_wmma_f32_16x16x32_bf16(false, a, false, bfrag,
                                                    (short)0, c, false, false);
    }

    const float bv = bias[ocol];                // lane-fixed column -> one bias
    v8bf ov;
#pragma unroll
    for (int r = 0; r < 8; ++r) {
        float v = c[r] + bv;
        v = v > 0.0f ? v : 0.0f;
        ov[r] = (__bf16)v;
    }
    // C lane holds column ocol, rows mbase..mbase+7 -> store transposed as
    // 8 contiguous bf16 (16B) into hT[b][o][n]
    const int mbase = mtile * 16 + half * 8;
    const int bidx  = mbase >> 8;               // /256
    const int nbase = mbase & 255;
    __bf16* dst = hT + ((size_t)bidx * OUTD + ocol) * NN + nbase;
    *(v8bf*)dst = ov;
}

// ---------------------------------------------------------------------------
// Stage 2: out[b][n][k*64+d] = sum_m adj[b][k][n][m] * hT[b][k*64+d][m]
// grid = 512 blocks (one per (b,k)), 8 waves; wave -> 2 n-tiles x 4 d-tiles.
// A-frag (adj, f32->bf16) reused across the 4 d-tile WMMAs each k-step.
// ---------------------------------------------------------------------------
__global__ void __launch_bounds__(256)
kgc_aggregate(const float* __restrict__ adj, const __bf16* __restrict__ hT,
              float* __restrict__ out) {
    const int lane = threadIdx.x & 31;
    const int wave = threadIdx.x >> 5;
    const int bk   = blockIdx.x;                // 0..511
    const int b    = bk >> 3;
    const int k    = bk & 7;
    const int half = lane >> 4;
    const int l15  = lane & 15;

    const float* A = adj + (size_t)bk * NN * NN;        // [n][m], stride 256

#pragma unroll
    for (int t = 0; t < 2; ++t) {
        const int ntile = wave * 2 + t;
        const int nrow  = ntile * 16 + l15;
        const float* arow = A + (size_t)nrow * NN + half * 16;

        v8f acc0 = {}, acc1 = {}, acc2 = {}, acc3 = {};
#pragma unroll 2
        for (int ks = 0; ks < NN / 32; ++ks) {
            v4f x0 = *(const v4f*)(arow + ks * 32);
            v4f x1 = *(const v4f*)(arow + ks * 32 + 4);
            v4f x2 = *(const v4f*)(arow + ks * 32 + 8);
            v4f x3 = *(const v4f*)(arow + ks * 32 + 12);
            v16bf a;
#pragma unroll
            for (int j = 0; j < 4; ++j) {
                a[j]      = (__bf16)x0[j];
                a[4 + j]  = (__bf16)x1[j];
                a[8 + j]  = (__bf16)x2[j];
                a[12 + j] = (__bf16)x3[j];
            }
            const __bf16* hbase =
                hT + ((size_t)b * OUTD + k * DKD + l15) * NN + ks * 32 + half * 16;
            // 4 d-tiles: columns o = k*64 + dt*16 + l15 (row stride NN in hT)
            v16bf b0 = *(const v16bf*)(hbase + (size_t)0  * NN);
            v16bf b1 = *(const v16bf*)(hbase + (size_t)16 * NN);
            v16bf b2 = *(const v16bf*)(hbase + (size_t)32 * NN);
            v16bf b3 = *(const v16bf*)(hbase + (size_t)48 * NN);
            acc0 = __builtin_amdgcn_wmma_f32_16x16x32_bf16(false, a, false, b0,
                                                           (short)0, acc0, false, false);
            acc1 = __builtin_amdgcn_wmma_f32_16x16x32_bf16(false, a, false, b1,
                                                           (short)0, acc1, false, false);
            acc2 = __builtin_amdgcn_wmma_f32_16x16x32_bf16(false, a, false, b2,
                                                           (short)0, acc2, false, false);
            acc3 = __builtin_amdgcn_wmma_f32_16x16x32_bf16(false, a, false, b3,
                                                           (short)0, acc3, false, false);
        }

        // Store: lane holds column d = k*64 + dt*16 + l15, rows nbase..nbase+7
        const int nbase = ntile * 16 + half * 8;
        float* obase = out + ((size_t)(b * NN + nbase)) * OUTD + k * DKD + l15;
        v8f accs[4] = {acc0, acc1, acc2, acc3};
#pragma unroll
        for (int dt = 0; dt < 4; ++dt) {
            float* ocol = obase + dt * 16;
#pragma unroll
            for (int r = 0; r < 8; ++r)
                ocol[(size_t)r * OUTD] = accs[dt][r];
        }
    }
}

// ---------------------------------------------------------------------------
extern "C" void kernel_launch(void* const* d_in, const int* in_sizes, int n_in,
                              void* d_out, int out_size, void* d_ws, size_t ws_size,
                              hipStream_t stream) {
    const float* node_feats = (const float*)d_in[0];  // (64,256,512)
    const float* adj        = (const float*)d_in[1];  // (64,8,256,256)
    const float* weight     = (const float*)d_in[2];  // (512,512)
    const float* bias       = (const float*)d_in[3];  // (512,)
    float* out = (float*)d_out;                       // (64,256,512)

    __bf16* wT = (__bf16*)d_ws;                                   // 512 KB
    __bf16* hT = (__bf16*)((char*)d_ws + (size_t)IND * OUTD * 2); // 16 MB

    // Stage 0: transpose+cvt weight
    kgc_transpose_w<<<(IND * OUTD) / 256, 256, 0, stream>>>(weight, wT);

    // Stage 1: fused linear + bias + relu, output transposed bf16
    dim3 g1((BB * NN) / 16, OUTD / (16 * 8));  // (1024, 4)
    kgc_linear_relu<<<g1, 256, 0, stream>>>(node_feats, wT, bias, hT);

    // Stage 2: per-(b,k) adjacency aggregation
    kgc_aggregate<<<BB * KK, 256, 0, stream>>>(adj, hT, out);
}
